// Block_6536940224809
// MI455X (gfx1250) — matrix operand order repaired and checked by reading, not compiled, not run on hardware
//
#include <hip/hip_runtime.h>
#include <math.h>

typedef __bf16 bhalf;
typedef __attribute__((ext_vector_type(16))) __bf16 v16bf;
typedef __attribute__((ext_vector_type(8)))  __bf16 v8bf;
typedef __attribute__((ext_vector_type(8)))  float  v8f;
typedef __attribute__((ext_vector_type(4)))  __bf16 v4bf;

#define BB   8
#define TT   2048
#define CC   1024
#define FF   4096
#define MM   (BB * TT)          // 16384 token rows

// ---------------------------------------------------------------------------
// fp32 -> bf16 conversion (weights, x)
// ---------------------------------------------------------------------------
__global__ void cvt_f32_bf16(const float* __restrict__ in, bhalf* __restrict__ out, long n) {
    long i = (long)blockIdx.x * blockDim.x + threadIdx.x;
    long stride = (long)gridDim.x * blockDim.x;
    for (; i < n; i += stride) out[i] = (bhalf)in[i];
}

// ---------------------------------------------------------------------------
// LayerNorm over C=1024, one 256-thread block per row, bf16 output
// ---------------------------------------------------------------------------
__global__ __launch_bounds__(256) void ln_kernel(const float* __restrict__ x,
                                                 const float* __restrict__ w,
                                                 const float* __restrict__ b,
                                                 bhalf* __restrict__ out) {
    const int tid = threadIdx.x;
    const size_t row = blockIdx.x;
    const float* xr = x + row * CC;
    float4 v = ((const float4*)xr)[tid];                 // 256 * 4 = 1024
    float s  = v.x + v.y + v.z + v.w;
    float s2 = v.x * v.x + v.y * v.y + v.z * v.z + v.w * v.w;
    __shared__ float ssum[256];
    __shared__ float ssq[256];
    ssum[tid] = s; ssq[tid] = s2;
    __syncthreads();
    for (int off = 128; off > 0; off >>= 1) {
        if (tid < off) { ssum[tid] += ssum[tid + off]; ssq[tid] += ssq[tid + off]; }
        __syncthreads();
    }
    float mu  = ssum[0] * (1.0f / CC);
    float var = ssq[0] * (1.0f / CC) - mu * mu;
    float rs  = rsqrtf(var + 1e-5f);
    float4 wv = ((const float4*)w)[tid];
    float4 bv = ((const float4*)b)[tid];
    v4bf o;
    o[0] = (bhalf)((v.x - mu) * rs * wv.x + bv.x);
    o[1] = (bhalf)((v.y - mu) * rs * wv.y + bv.y);
    o[2] = (bhalf)((v.z - mu) * rs * wv.z + bv.z);
    o[3] = (bhalf)((v.w - mu) * rs * wv.w + bv.w);
    *(v4bf*)(out + row * CC + tid * 4) = o;
}

// ---------------------------------------------------------------------------
// Time-shift mixing: o = h[t]*mix + h[t-1]*(1-mix), up to 3 outputs
// ---------------------------------------------------------------------------
__global__ void mix_kernel(const bhalf* __restrict__ h,
                           const float* __restrict__ m0, const float* __restrict__ m1,
                           const float* __restrict__ m2,
                           bhalf* __restrict__ o0, bhalf* __restrict__ o1,
                           bhalf* __restrict__ o2, long total) {
    long i = (long)blockIdx.x * blockDim.x + threadIdx.x;
    long stride = (long)gridDim.x * blockDim.x;
    for (; i < total; i += stride) {
        int  c  = (int)(i & (CC - 1));
        long bt = i >> 10;                    // / CC
        int  t  = (int)(bt & (TT - 1));
        float cur  = (float)h[i];
        float prev = (t == 0) ? 0.0f : (float)h[i - CC];
        float mk = m0[c];
        o0[i] = (bhalf)(cur * mk + prev * (1.0f - mk));
        if (o1) { float mv = m1[c]; o1[i] = (bhalf)(cur * mv + prev * (1.0f - mv)); }
        if (o2) { float mr = m2[c]; o2[i] = (bhalf)(cur * mr + prev * (1.0f - mr)); }
    }
}

// ---------------------------------------------------------------------------
// WKV recurrence: one thread per (b,c) channel, scan over T (coalesced in C)
// ---------------------------------------------------------------------------
__global__ void wkv_kernel(const float* __restrict__ dec, const float* __restrict__ first,
                           const float* __restrict__ k, const float* __restrict__ v,
                           float* __restrict__ y) {
    int gid = blockIdx.x * blockDim.x + threadIdx.x;
    if (gid >= BB * CC) return;
    int c = gid & (CC - 1);
    int b = gid >> 10;
    float w = -expf(dec[c]);
    float u = first[c];
    float aa = 0.0f, bb = 0.0f, pp = -1e30f;
    size_t base = (size_t)b * TT * CC + c;
    for (int t = 0; t < TT; ++t) {
        size_t i = base + (size_t)t * CC;
        float kt = k[i], vt = v[i];
        float ww = u + kt;
        float q  = fmaxf(pp, ww);
        float e1 = expf(pp - q), e2 = expf(ww - q);
        y[i] = (e1 * aa + e2 * vt) / (e1 * bb + e2);
        float ww2 = pp + w;
        float q2  = fmaxf(ww2, kt);
        float f1 = expf(ww2 - q2), f2 = expf(kt - q2);
        aa = f1 * aa + f2 * vt;
        bb = f1 * bb + f2;
        pp = q2;
    }
}

// ---------------------------------------------------------------------------
// rwkv = sr * wkv   (bf16 out)
// ---------------------------------------------------------------------------
__global__ void mul_sr_kernel(const bhalf* __restrict__ sr, const float* __restrict__ wkv,
                              bhalf* __restrict__ out, long n) {
    long i = (long)blockIdx.x * blockDim.x + threadIdx.x;
    long stride = (long)gridDim.x * blockDim.x;
    for (; i < n; i += stride) out[i] = (bhalf)((float)sr[i] * wkv[i]);
}

// ---------------------------------------------------------------------------
// out = xmid + fr * kv   (f32 out; fr already sigmoid'ed, bf16)
// ---------------------------------------------------------------------------
__global__ void final_kernel(const float* __restrict__ xmid, const bhalf* __restrict__ fr,
                             const float* __restrict__ kv, float* __restrict__ out, long n) {
    long i = (long)blockIdx.x * blockDim.x + threadIdx.x;
    long stride = (long)gridDim.x * blockDim.x;
    for (; i < n; i += stride) out[i] = xmid[i] + (float)fr[i] * kv[i];
}

// ---------------------------------------------------------------------------
// CDNA5 async copy: global -> LDS, 16B per lane, tracked by ASYNCcnt.
// Low 32 bits of a generic shared pointer are the LDS byte offset (ISA 10.2).
// Async loads complete IN ORDER, so a threshold wait proves the oldest
// tile-group (6 ops/thread) has fully landed.
// ---------------------------------------------------------------------------
__device__ __forceinline__ void async_copy16(void* lds_ptr, const void* gptr) {
    unsigned lds_off = (unsigned)(unsigned long long)lds_ptr;
    asm volatile("global_load_async_to_lds_b128 %0, %1, off"
                 :: "v"(lds_off), "v"(gptr) : "memory");
}
__device__ __forceinline__ void wait_async0() {
    asm volatile("s_wait_asynccnt 0x0" ::: "memory");
}
__device__ __forceinline__ void wait_async6() {
    asm volatile("s_wait_asynccnt 0x6" ::: "memory");
}

// ---------------------------------------------------------------------------
// bf16 WMMA GEMM: out[m,n] = sum_k A[m,k] * W[n,k]   (W row-major [N,K])
// 256 threads = 8 waves (4x2), block tile 256x128, wave tile 64x64, BK=32.
// Triple-buffered LDS fed by global_load_async_to_lds_b128 with two tile
// groups in flight (ASYNCcnt threshold waits) -> ~2 compute phases of
// memory-latency tolerance per tile.
// EPI: 0 = store f32, 1 = f32 + addend, 2 = sigmoid->bf16, 3 = relu^2 ->bf16
// ---------------------------------------------------------------------------
#define BM  256
#define BN  128
#define BK  32
#define LDT 40          // padded LDS row stride (bf16 elems): 80B -> bank spread

union FragU { v16bf v; v8bf h[2]; };

template <int EPI>
__global__ __launch_bounds__(256) void gemm_bf16_wmma(const bhalf* __restrict__ A,
                                                      const bhalf* __restrict__ Bw,
                                                      void* __restrict__ Out,
                                                      const float* __restrict__ Addend,
                                                      int N, int K) {
    __shared__ __align__(16) bhalf As[3][BM * LDT];   // 3 x 20 KiB
    __shared__ __align__(16) bhalf Bs[3][BN * LDT];   // 3 x 10 KiB

    const int tid  = threadIdx.x;
    const int lane = tid & 31;
    const int w    = tid >> 5;
    const int wm   = w >> 1;          // 0..3  -> 64-row slice
    const int wn   = w & 1;           // 0..1  -> 64-col slice
    const int lm   = lane & 15;
    const int sel  = lane >> 4;       // lane half

    const size_t tileM = (size_t)blockIdx.x * BM;
    const size_t tileN = (size_t)blockIdx.y * BN;

    // async tile feeder: A = 1024 16B-chunks (4/thread), B = 512 (2/thread)
    auto issue_tiles = [&](int k0, int buf) {
#pragma unroll
        for (int j = 0; j < 4; ++j) {
            int c = j * 256 + tid, row = c >> 2, q = c & 3;
            async_copy16(&As[buf][row * LDT + q * 8],
                         A + (tileM + row) * (size_t)K + k0 + q * 8);
        }
#pragma unroll
        for (int j = 0; j < 2; ++j) {
            int c = j * 256 + tid, row = c >> 2, q = c & 3;
            async_copy16(&Bs[buf][row * LDT + q * 8],
                         Bw + (tileN + row) * (size_t)K + k0 + q * 8);
        }
    };

    v8f acc[4][4] = {};

    issue_tiles(0, 0);
    if (BK < K) issue_tiles(BK, 1);

    int cur = 0;
    for (int k0 = 0; k0 < K; k0 += BK) {
        // tile 'cur' is the oldest in flight: 6 ops behind the (possible)
        // next tile's 6 -> threshold 6 proves it landed; last tile: drain.
        if (k0 + BK < K) wait_async6(); else wait_async0();
        __syncthreads();     // all waves: tile cur ready, prev buf reads done

        int nb = cur + 2; if (nb >= 3) nb -= 3;
        if (k0 + 2 * BK < K) issue_tiles(k0 + 2 * BK, nb);   // stream tile i+2

        // B fragments once (32 VGPRs live), then per-mi A fragment + 4 WMMAs
        FragU bfr[4];
#pragma unroll
        for (int ni = 0; ni < 4; ++ni) {
            const bhalf* rp = &Bs[cur][(wn * 64 + ni * 16 + lm) * LDT];
            bfr[ni].h[0] = *(const v8bf*)(rp + sel * 8);         // K = sel*8..+7
            bfr[ni].h[1] = *(const v8bf*)(rp + 16 + sel * 8);    // K = 16+sel*8..+7
        }
#pragma unroll
        for (int mi = 0; mi < 4; ++mi) {
            FragU af;
            const bhalf* rp = &As[cur][(wm * 64 + mi * 16 + lm) * LDT];
            af.h[0] = *(const v8bf*)(rp + sel * 8);              // K = sel*8..+7
            af.h[1] = *(const v8bf*)(rp + 16 + sel * 8);         // K = 16+sel*8..+7
#pragma unroll
            for (int ni = 0; ni < 4; ++ni)
                acc[mi][ni] = __builtin_amdgcn_wmma_f32_16x16x32_bf16(
                    false, af.v, false, bfr[ni].v, (short)0, acc[mi][ni], false, false);
        }

        cur = (cur + 1 == 3) ? 0 : cur + 1;
    }

    // Epilogue: lane L, vgpr r -> M = r + (L>>4)*8, N = L&15   (ISA 7.12.2)
#pragma unroll
    for (int mi = 0; mi < 4; ++mi) {
#pragma unroll
        for (int ni = 0; ni < 4; ++ni) {
            size_t n     = tileN + wn * 64 + ni * 16 + lm;
            size_t mbase = tileM + wm * 64 + mi * 16 + sel * 8;
#pragma unroll
            for (int r = 0; r < 8; ++r) {
                float  val = acc[mi][ni][r];
                size_t idx = (mbase + r) * (size_t)N + n;
                if (EPI == 0) {
                    ((float*)Out)[idx] = val;
                } else if (EPI == 1) {
                    ((float*)Out)[idx] = val + Addend[idx];
                } else if (EPI == 2) {
                    ((bhalf*)Out)[idx] = (bhalf)(1.0f / (1.0f + expf(-val)));
                } else {
                    float t = fmaxf(val, 0.0f);
                    ((bhalf*)Out)[idx] = (bhalf)(t * t);
                }
            }
        }
    }
}

// ---------------------------------------------------------------------------
// Host orchestration
// ---------------------------------------------------------------------------
extern "C" void kernel_launch(void* const* d_in, const int* in_sizes, int n_in,
                              void* d_out, int out_size, void* d_ws, size_t ws_size,
                              hipStream_t stream) {
    const float* x      = (const float*)d_in[0];
    const float* ln1_w  = (const float*)d_in[1];
    const float* ln1_b  = (const float*)d_in[2];
    const float* ln2_w  = (const float*)d_in[3];
    const float* ln2_b  = (const float*)d_in[4];
    const float* decay  = (const float*)d_in[5];
    const float* first  = (const float*)d_in[6];
    const float* mix_k  = (const float*)d_in[7];
    const float* mix_v  = (const float*)d_in[8];
    const float* mix_r  = (const float*)d_in[9];
    const float* Wk     = (const float*)d_in[10];
    const float* Wv     = (const float*)d_in[11];
    const float* Wr     = (const float*)d_in[12];
    const float* Wo     = (const float*)d_in[13];
    const float* fmix_k = (const float*)d_in[14];
    const float* fmix_r = (const float*)d_in[15];
    const float* fWk    = (const float*)d_in[16];   // [F,C]
    const float* fWr    = (const float*)d_in[17];   // [C,C]
    const float* fWv    = (const float*)d_in[18];   // [C,F]
    const float* shW    = (const float*)d_in[19];   // [C,C]
    float* out = (float*)d_out;

    char* ws = (char*)d_ws;
    const size_t MiB = 1ull << 20;
    // bf16 weights (persistent)
    bhalf* wkb  = (bhalf*)(ws + 0 * MiB);
    bhalf* wvb  = (bhalf*)(ws + 2 * MiB);
    bhalf* wrb  = (bhalf*)(ws + 4 * MiB);
    bhalf* wob  = (bhalf*)(ws + 6 * MiB);
    bhalf* wsb  = (bhalf*)(ws + 8 * MiB);
    bhalf* fwkb = (bhalf*)(ws + 10 * MiB);  // 8 MiB
    bhalf* fwrb = (bhalf*)(ws + 18 * MiB);
    bhalf* fwvb = (bhalf*)(ws + 20 * MiB);  // 8 MiB
    // activation pool with lifetime reuse
    bhalf* xk   = (bhalf*)(ws + 32 * MiB);   // 32 MiB
    bhalf* xv   = (bhalf*)(ws + 64 * MiB);   // 32 MiB
    bhalf* xr   = (bhalf*)(ws + 96 * MiB);   // 32 MiB
    bhalf* h    = (bhalf*)(ws + 128 * MiB);  // 32 MiB
    float* kbuf = (float*)(ws + 160 * MiB);  // 64 MiB
    float* vbuf = (float*)(ws + 224 * MiB);  // 64 MiB
    bhalf* sr   = (bhalf*)(ws + 128 * MiB);  // reuse h (dead after mix)
    float* wkvb = (float*)(ws + 32 * MiB);   // reuse xk+xv (dead after k,v GEMMs)
    bhalf* rwkv = (bhalf*)(ws + 96 * MiB);   // reuse xr (dead after r GEMM)
    float* att  = (float*)(ws + 160 * MiB);  // reuse kbuf (dead after wkv)
    float* xmid = (float*)(ws + 224 * MiB);  // reuse vbuf (dead after wkv)
    bhalf* g    = (bhalf*)(ws + 32 * MiB);   // reuse (wkv dead after rwkv)
    bhalf* gk   = (bhalf*)(ws + 64 * MiB);
    bhalf* gr   = (bhalf*)(ws + 128 * MiB);  // reuse sr (dead after rwkv)
    bhalf* fr   = (bhalf*)(ws + 96 * MiB);   // reuse rwkv (dead after att GEMM)
    float* kv   = (float*)(ws + 160 * MiB);  // reuse att (dead after xmid GEMM)
    bhalf* kk   = (bhalf*)(ws + 288 * MiB);  // 128 MiB
    bhalf* xbf  = (bhalf*)(ws + 416 * MiB);  // 32 MiB

    const long MC = (long)MM * CC;

    // 1. weight / input conversions to bf16
    cvt_f32_bf16<<<2048, 256, 0, stream>>>(Wk,  wkb,  (long)CC * CC);
    cvt_f32_bf16<<<2048, 256, 0, stream>>>(Wv,  wvb,  (long)CC * CC);
    cvt_f32_bf16<<<2048, 256, 0, stream>>>(Wr,  wrb,  (long)CC * CC);
    cvt_f32_bf16<<<2048, 256, 0, stream>>>(Wo,  wob,  (long)CC * CC);
    cvt_f32_bf16<<<2048, 256, 0, stream>>>(shW, wsb,  (long)CC * CC);
    cvt_f32_bf16<<<2048, 256, 0, stream>>>(fWk, fwkb, (long)FF * CC);
    cvt_f32_bf16<<<2048, 256, 0, stream>>>(fWr, fwrb, (long)CC * CC);
    cvt_f32_bf16<<<2048, 256, 0, stream>>>(fWv, fwvb, (long)CC * FF);
    cvt_f32_bf16<<<4096, 256, 0, stream>>>(x,   xbf,  MC);

    // 2. TimeMix path
    ln_kernel<<<MM, 256, 0, stream>>>(x, ln1_w, ln1_b, h);
    mix_kernel<<<8192, 256, 0, stream>>>(h, mix_k, mix_v, mix_r, xk, xv, xr, MC);

    dim3 gcc(MM / BM, CC / BN);     // 64 x 8
    dim3 gfc(MM / BM, FF / BN);     // 64 x 32
    gemm_bf16_wmma<0><<<gcc, 256, 0, stream>>>(xk, wkb, (void*)kbuf, nullptr, CC, CC);
    gemm_bf16_wmma<0><<<gcc, 256, 0, stream>>>(xv, wvb, (void*)vbuf, nullptr, CC, CC);
    gemm_bf16_wmma<2><<<gcc, 256, 0, stream>>>(xr, wrb, (void*)sr,   nullptr, CC, CC);

    wkv_kernel<<<(BB * CC) / 256, 256, 0, stream>>>(decay, first, kbuf, vbuf, wkvb);
    mul_sr_kernel<<<8192, 256, 0, stream>>>(sr, wkvb, rwkv, MC);

    gemm_bf16_wmma<0><<<gcc, 256, 0, stream>>>(rwkv, wob, (void*)att, nullptr, CC, CC);
    gemm_bf16_wmma<1><<<gcc, 256, 0, stream>>>(xbf,  wsb, (void*)xmid, att,    CC, CC);

    // 3. ChannelMix path
    ln_kernel<<<MM, 256, 0, stream>>>(xmid, ln2_w, ln2_b, g);
    mix_kernel<<<8192, 256, 0, stream>>>(g, fmix_k, fmix_r, nullptr, gk, gr, nullptr, MC);

    gemm_bf16_wmma<3><<<gfc, 256, 0, stream>>>(gk, fwkb, (void*)kk, nullptr, FF, CC);
    gemm_bf16_wmma<2><<<gcc, 256, 0, stream>>>(gr, fwrb, (void*)fr, nullptr, CC, CC);
    gemm_bf16_wmma<0><<<gcc, 256, 0, stream>>>(kk, fwvb, (void*)kv, nullptr, CC, FF);

    final_kernel<<<8192, 256, 0, stream>>>(xmid, fr, kv, out, MC);
}
